// BatchHoppy_63153199120385
// MI455X (gfx1250) — matrix operand order repaired and checked by reading, not compiled, not run on hardware
//
#include <hip/hip_runtime.h>
#include <hip/hip_bf16.h>

// ---------------------------------------------------------------------------
// BatchHoppy depth-1 prover for MI455X (gfx1250, wave32).
// Heavy work = 2 batched [1024 x 2048, K=256] Gaussian-kernel GEMMs
// (entities vs facts) -> v_wmma_f32_16x16x32_bf16, A-frags register-resident,
// fact tiles double-buffered in LDS via global_load_async_to_lds_b128, and
// B-fragments double-buffered in registers to avoid full-DS-drain waits.
// ---------------------------------------------------------------------------

#define B_   16
#define E_   256
#define N_   1024
#define F_   2048

typedef __bf16 bhalf;
typedef bhalf  bhalf16 __attribute__((ext_vector_type(16)));
typedef float  v8f     __attribute__((ext_vector_type(8)));

union FragAB { bhalf16 v; unsigned int u[8]; };

// ---------------- kernel 1: hop_rel[r][h][b][:] = rel[b] @ W[r,h] + bias ----
__global__ __launch_bounds__(256) void k_hoprel(const float* __restrict__ rel,
                                                const float* __restrict__ W,
                                                const float* __restrict__ bias,
                                                float* __restrict__ hop) {
  int gid = blockIdx.x * 256 + threadIdx.x;     // 16384 total
  int eo = gid & 255;
  int b  = (gid >> 8) & 15;
  int rh = gid >> 12;                           // r*2+h, 0..3
  const float* Wp = W + (size_t)rh * E_ * E_ + eo;
  const float* rp = rel + b * E_;
  float acc = bias[rh * E_ + eo];
  for (int e = 0; e < E_; ++e) acc += rp[e] * Wp[(size_t)e * E_];
  hop[((size_t)rh * B_ + b) * E_ + eo] = acc;
}

// ---------------- kernel 2: f32 -> bf16 (RNE) + row squared-norms ----------
__global__ __launch_bounds__(256) void k_cvt(const float* __restrict__ src,
                                             unsigned short* __restrict__ dst,
                                             float* __restrict__ norms) {
  __shared__ float red[256];
  int row = blockIdx.x, tid = threadIdx.x;
  float x = src[(size_t)row * E_ + tid];
  unsigned int bits = __float_as_uint(x);
  unsigned int h = (bits + 0x7FFFu + ((bits >> 16) & 1u)) >> 16;
  dst[(size_t)row * E_ + tid] = (unsigned short)h;
  red[tid] = x * x;
  __syncthreads();
  for (int s = 128; s > 0; s >>= 1) {
    if (tid < s) red[tid] += red[tid + s];
    __syncthreads();
  }
  if (tid == 0) norms[row] = red[0];
}

// ---------------- kernel 3: nine small [B,F] kernel maps -> prefactors -----
__global__ __launch_bounds__(256) void k_const(const float* __restrict__ rel,
                                               const float* __restrict__ arg1,
                                               const float* __restrict__ arg2,
                                               const float* __restrict__ fr,
                                               const float* __restrict__ f1,
                                               const float* __restrict__ f2,
                                               const float* __restrict__ hop,
                                               const int*  __restrict__ nbf,
                                               float* __restrict__ c1,
                                               float* __restrict__ c2,
                                               float* __restrict__ s0) {
  __shared__ float q[7 * E_];
  int tid = threadIdx.x, lane = tid & 31, wave = tid >> 5;
  int b = blockIdx.y;
  q[0 * E_ + tid] = rel [b * E_ + tid];
  q[1 * E_ + tid] = arg1[b * E_ + tid];
  q[2 * E_ + tid] = arg2[b * E_ + tid];
  q[3 * E_ + tid] = hop[((size_t)0 * B_ + b) * E_ + tid];   // r0 h0
  q[4 * E_ + tid] = hop[((size_t)1 * B_ + b) * E_ + tid];   // r0 h1
  q[5 * E_ + tid] = hop[((size_t)2 * B_ + b) * E_ + tid];   // r1 h0
  q[6 * E_ + tid] = hop[((size_t)3 * B_ + b) * E_ + tid];   // r1 h1
  __syncthreads();
  int f = blockIdx.x * 8 + wave;
  const float* frb = fr + ((size_t)(b * F_ + f)) * E_;
  const float* f1b = f1 + ((size_t)(b * F_ + f)) * E_;
  const float* f2b = f2 + ((size_t)(b * F_ + f)) * E_;
  float d[9] = {0,0,0,0,0,0,0,0,0};
  for (int i = 0; i < 8; ++i) {
    int e = lane + i * 32;
    float vr = frb[e], v1 = f1b[e], v2 = f2b[e], t;
    t = q[0*E_+e] - vr; d[0] += t*t;   // rel  vs fact_rel
    t = q[3*E_+e] - vr; d[1] += t*t;   // h00  vs fact_rel
    t = q[4*E_+e] - vr; d[2] += t*t;   // h01  vs fact_rel
    t = q[5*E_+e] - vr; d[3] += t*t;   // h10  vs fact_rel
    t = q[6*E_+e] - vr; d[4] += t*t;   // h11  vs fact_rel
    t = q[1*E_+e] - v1; d[5] += t*t;   // arg1 vs fact_arg1
    t = q[2*E_+e] - v1; d[6] += t*t;   // arg2 vs fact_arg1
    t = q[1*E_+e] - v2; d[7] += t*t;   // arg1 vs fact_arg2
    t = q[2*E_+e] - v2; d[8] += t*t;   // arg2 vs fact_arg2
  }
  for (int j = 0; j < 9; ++j)
    for (int s = 16; s >= 1; s >>= 1) d[j] += __shfl_xor(d[j], s);
  if (lane == 0) {
    float m = (f < nbf[b]) ? 1.f : 0.f;
    float kr   = __expf(-0.5f * d[0]);
    float kh00 = __expf(-0.5f * d[1]);
    float kh01 = __expf(-0.5f * d[2]);
    float kh10 = __expf(-0.5f * d[3]);
    float kh11 = __expf(-0.5f * d[4]);
    float a1f1 = __expf(-0.5f * d[5]);
    float a2f1 = __expf(-0.5f * d[6]);
    float a1f2 = __expf(-0.5f * d[7]);
    float a2f2 = __expf(-0.5f * d[8]);
    size_t base = (size_t)b * F_ + f;
    c1[0 * (size_t)B_ * F_ + base] = m * kh00 * a1f1;  // rule0 hop1 const
    c1[1 * (size_t)B_ * F_ + base] = m * kh10 * a1f2;  // rule1 hop1 const
    c2[0 * (size_t)B_ * F_ + base] = m * kh01 * a2f2;  // rule0 hop2 const
    c2[1 * (size_t)B_ * F_ + base] = m * kh11 * a2f1;  // rule1 hop2 const
    s0[base] = m * kr * a1f1 * a2f2;                    // depth-0 product
  }
}

// ---------------- kernel 4: big WMMA hop-1 kernel ---------------------------
// new_scores[r][b][n] = max_f c1[r][b][f] * exp(-0.5*max(en+fn-2*ent.fact,0))
// Phase 1: stage 64x256 bf16 entity tile -> LDS -> register A-fragments.
// Phase 2: reuse LDS as double-buffered 32-fact tiles filled with
//          global_load_async_to_lds_b128 (ASYNCcnt), overlapped with WMMA.
//          B-fragments are double-buffered in registers so DS loads of
//          k-step ks+1 overlap the WMMAs of k-step ks (partial dscnt waits).
#define STR_  132                 // dwords per LDS row (128 data + 4 pad)
#define FBUF_ (32 * STR_)         // one 32-fact buffer, in dwords
__global__ __launch_bounds__(128) void k_big(const unsigned short* __restrict__ entb,
                                             const unsigned short* __restrict__ fa1b,
                                             const unsigned short* __restrict__ fa2b,
                                             const float* __restrict__ en,
                                             const float* __restrict__ fn1,
                                             const float* __restrict__ fn2,
                                             const float* __restrict__ c1,
                                             float* __restrict__ nsc) {
  __shared__ unsigned int lds[2 * FBUF_];   // 33792 B; phase1 uses [0, 64*STR_)
  const int tid  = threadIdx.x;
  const int lane = tid & 31;
  const int wave = tid >> 5;
  const int n0   = blockIdx.x * 64;
  const int b    = blockIdx.y;
  const int r    = blockIdx.z;
  const unsigned short* factb = r ? fa1b : fa2b;   // rule0: fact_arg2, rule1: fact_arg1
  const float* fn   = r ? fn1 : fn2;
  const float* cfac = c1 + ((size_t)(r * B_ + b)) * F_;

  const int mloc  = lane & 15;
  const int khA   = (lane < 16) ? 0 : 16;     // A-frag K half-offset
  const int koffB = (lane < 16) ? 0 : 16;     // B-frag K half-offset
  const int m0    = wave * 16;

  // ---- phase 1: entity tile -> LDS -> A fragments in registers ----
  const uint4* entg = reinterpret_cast<const uint4*>(entb) +
                      ((size_t)(b * N_ + n0)) * 32;          // 32 uint4 / row
  for (int t = 0; t < 16; ++t) {
    int idx = tid + t * 128;                  // 0..2047 (64 rows x 32 uint4)
    int row = idx >> 5, c4 = idx & 31;
    *reinterpret_cast<uint4*>(&lds[row * STR_ + c4 * 4]) = entg[row * 32 + c4];
  }
  __syncthreads();
  FragAB afr[8];
  {
    const unsigned int* Arow = &lds[(m0 + mloc) * STR_];
#pragma unroll
    for (int ks = 0; ks < 8; ++ks) {
      int k0 = ks * 32;
#pragma unroll
      for (int j = 0; j < 4; ++j) afr[ks].u[j]     = Arow[(k0 + khA + 2 * j) >> 1];
#pragma unroll
      for (int j = 0; j < 4; ++j) afr[ks].u[4 + j] = Arow[(k0 + khA + 8 + 2 * j) >> 1];
    }
  }
  float enreg[8], runm[8];
#pragma unroll
  for (int v = 0; v < 8; ++v) {
    int nrow = n0 + m0 + v + ((lane < 16) ? 0 : 8);
    enreg[v] = en[b * N_ + nrow];
    runm[v]  = 0.f;
  }
  __syncthreads();    // phase-1 LDS reads done before fact staging overwrites

  // ---- phase 2: f-loop, 32 facts/iter, async double-buffered staging ----
  const int NT = F_ / 32;                     // 64 tiles
#define STAGE(ft, buf)                                                         \
  {                                                                            \
    const uint4* fg = reinterpret_cast<const uint4*>(factb) +                  \
                      ((size_t)(b * F_ + (ft) * 32)) * 32;                     \
    _Pragma("unroll")                                                          \
    for (int t = 0; t < 8; ++t) {                                              \
      int idx = tid + t * 128;               /* 0..1023 */                     \
      int row = idx >> 5, c4 = idx & 31;                                       \
      unsigned lad = (unsigned)(uintptr_t)&lds[(buf) * FBUF_ + row * STR_ + c4 * 4]; \
      unsigned long long ga = (unsigned long long)(uintptr_t)(fg + row * 32 + c4);   \
      asm volatile("global_load_async_to_lds_b128 %0, %1, off"                 \
                   :: "v"(lad), "v"(ga) : "memory");                           \
    }                                                                          \
  }

  STAGE(0, 0);
  asm volatile("s_wait_asynccnt 0" ::: "memory");
  __syncthreads();

  for (int ft = 0; ft < NT; ++ft) {
    const int cur = ft & 1;
    if (ft + 1 < NT) STAGE(ft + 1, 1 - cur);   // overlaps with compute below

    v8f acc0 = {}, acc1 = {};
    const unsigned int* B0 = &lds[cur * FBUF_ + mloc * STR_];
    const unsigned int* B1 = &lds[cur * FBUF_ + (16 + mloc) * STR_];

    // register-double-buffered B fragments: loads for k-step ks+1 are issued
    // before the WMMAs consuming k-step ks, enabling partial dscnt waits.
    FragAB bb0[2], bb1[2];
#pragma unroll
    for (int j = 0; j < 8; ++j) bb0[0].u[j] = B0[(koffB + 2 * j) >> 1];
#pragma unroll
    for (int j = 0; j < 8; ++j) bb1[0].u[j] = B1[(koffB + 2 * j) >> 1];
#pragma unroll
    for (int ks = 0; ks < 8; ++ks) {
      const int pb = ks & 1, nb = pb ^ 1;
      if (ks + 1 < 8) {
        int k0n = (ks + 1) * 32;
#pragma unroll
        for (int j = 0; j < 8; ++j) bb0[nb].u[j] = B0[(k0n + koffB + 2 * j) >> 1];
#pragma unroll
        for (int j = 0; j < 8; ++j) bb1[nb].u[j] = B1[(k0n + koffB + 2 * j) >> 1];
      }
      acc0 = __builtin_amdgcn_wmma_f32_16x16x32_bf16(false, afr[ks].v, false, bb0[pb].v,
                                                     (short)0, acc0, false, false);
      acc1 = __builtin_amdgcn_wmma_f32_16x16x32_bf16(false, afr[ks].v, false, bb1[pb].v,
                                                     (short)0, acc1, false, false);
    }

    int f0 = ft * 32 + mloc;
    int f1 = f0 + 16;
    float fnv0 = fn[b * F_ + f0], cv0 = cfac[f0];
    float fnv1 = fn[b * F_ + f1], cv1 = cfac[f1];
#pragma unroll
    for (int v = 0; v < 8; ++v) {
      float d0 = fmaxf(enreg[v] + fnv0 - 2.f * acc0[v], 0.f);
      float d1 = fmaxf(enreg[v] + fnv1 - 2.f * acc1[v], 0.f);
      runm[v] = fmaxf(runm[v], cv0 * __expf(-0.5f * d0));
      runm[v] = fmaxf(runm[v], cv1 * __expf(-0.5f * d1));
    }

    if (ft + 1 < NT) asm volatile("s_wait_asynccnt 0" ::: "memory");
    __syncthreads();
  }
#undef STAGE

  // max across the 16 lanes holding different f within each half
#pragma unroll
  for (int v = 0; v < 8; ++v) {
    float m = runm[v];
    for (int s = 8; s >= 1; s >>= 1) m = fmaxf(m, __shfl_xor(m, s));
    runm[v] = m;
  }
  float* outrow = nsc + ((size_t)(r * B_ + b)) * N_ + n0 + m0;
  if (lane == 0)
    for (int v = 0; v < 8; ++v) outrow[v] = runm[v];
  if (lane == 16)
    for (int v = 0; v < 8; ++v) outrow[8 + v] = runm[v];
}

// ---------------- kernel 5: top-10 per (rule,b) over N ----------------------
__global__ __launch_bounds__(256) void k_topk(const float* __restrict__ nsc,
                                              float* __restrict__ zs,
                                              int*   __restrict__ zi) {
  __shared__ float sv[N_];
  __shared__ float rv[256];
  __shared__ int   ri[256];
  int rb = blockIdx.x, tid = threadIdx.x;
  const float* row = nsc + (size_t)rb * N_;
  for (int j = 0; j < 4; ++j) sv[tid + 256 * j] = row[tid + 256 * j];
  __syncthreads();
  for (int it = 0; it < 10; ++it) {
    float best = -1e30f; int bi = 0;
    for (int j = 0; j < 4; ++j) {
      int i = tid * 4 + j;
      float v = sv[i];
      if (v > best) { best = v; bi = i; }
    }
    rv[tid] = best; ri[tid] = bi;
    __syncthreads();
    for (int s = 128; s > 0; s >>= 1) {
      if (tid < s && rv[tid + s] > rv[tid]) { rv[tid] = rv[tid + s]; ri[tid] = ri[tid + s]; }
      __syncthreads();
    }
    if (tid == 0) {
      zs[rb * 10 + it] = rv[0];
      zi[rb * 10 + it] = ri[0];
      sv[ri[0]] = -1e30f;
    }
    __syncthreads();
  }
}

// ---------------- kernel 6: hop-2 score per (r,b,branch) -------------------
__global__ __launch_bounds__(256) void k_hop2(const float* __restrict__ ent,
                                              const float* __restrict__ f1,
                                              const float* __restrict__ f2,
                                              const float* __restrict__ c2,
                                              const float* __restrict__ zs,
                                              const int*   __restrict__ zi,
                                              float* __restrict__ branch) {
  __shared__ float src[E_];
  __shared__ float wm[8];
  int tid = threadIdx.x, lane = tid & 31, wave = tid >> 5;
  int idx = blockIdx.x;           // 0..319
  int j = idx % 10, b = (idx / 10) & 15, r = idx / 160;
  int si = zi[(r * B_ + b) * 10 + j];
  src[tid] = ent[((size_t)(b * N_ + si)) * E_ + tid];
  __syncthreads();
  const float* facts = r ? f2 : f1;   // rule0: src vs fact_arg1, rule1: src vs fact_arg2
  const float* cf = c2 + ((size_t)(r * B_ + b)) * F_;
  float rm = 0.f;
  for (int i = 0; i < 256; ++i) {
    int f = wave * 256 + i;
    const float* fb = facts + ((size_t)(b * F_ + f)) * E_;
    float d = 0.f;
    for (int e = lane; e < E_; e += 32) { float t = src[e] - fb[e]; d += t * t; }
    for (int s = 16; s >= 1; s >>= 1) d += __shfl_xor(d, s);
    rm = fmaxf(rm, cf[f] * __expf(-0.5f * d));
  }
  if (lane == 0) wm[wave] = rm;
  __syncthreads();
  if (tid == 0) {
    float m = wm[0];
    for (int w = 1; w < 8; ++w) m = fmaxf(m, wm[w]);
    float z = zs[(r * B_ + b) * 10 + j];
    branch[(r * B_ + b) * 10 + j] = fminf(m, z);   // min(hop2, hop1 top-k)
  }
}

// ---------------- kernel 7: final combine ----------------------------------
__global__ __launch_bounds__(256) void k_final(const float* __restrict__ s0,
                                               const float* __restrict__ branch,
                                               float* __restrict__ out) {
  __shared__ float red[256];
  int b = blockIdx.x, tid = threadIdx.x;
  float m = 0.f;
  for (int j = 0; j < 8; ++j) m = fmaxf(m, s0[(size_t)b * F_ + tid + 256 * j]);
  red[tid] = m;
  __syncthreads();
  for (int s = 128; s > 0; s >>= 1) {
    if (tid < s) red[tid] = fmaxf(red[tid], red[tid + s]);
    __syncthreads();
  }
  if (tid == 0) {
    float g = -1e30f;
    for (int r = 0; r < 2; ++r)
      for (int j = 0; j < 10; ++j) g = fmaxf(g, branch[(r * B_ + b) * 10 + j]);
    out[b] = fmaxf(red[0], g);
  }
}

// ---------------------------------------------------------------------------
extern "C" void kernel_launch(void* const* d_in, const int* in_sizes, int n_in,
                              void* d_out, int out_size, void* d_ws, size_t ws_size,
                              hipStream_t stream) {
  const float* rel   = (const float*)d_in[0];
  const float* arg1  = (const float*)d_in[1];
  const float* arg2  = (const float*)d_in[2];
  const float* frel  = (const float*)d_in[3];
  const float* farg1 = (const float*)d_in[4];
  const float* farg2 = (const float*)d_in[5];
  const int*   nbf   = (const int*)d_in[6];
  const float* ent   = (const float*)d_in[7];
  const float* W     = (const float*)d_in[9];
  const float* bias  = (const float*)d_in[10];
  float* out = (float*)d_out;

  char* w = (char*)d_ws;
  size_t off = 0;
  auto alloc = [&](size_t bytes) -> void* {
    off = (off + 255) & ~(size_t)255;
    void* p = w + off;
    off += bytes;
    return p;
  };
  unsigned short* fa1b = (unsigned short*)alloc((size_t)B_ * F_ * E_ * 2);
  unsigned short* fa2b = (unsigned short*)alloc((size_t)B_ * F_ * E_ * 2);
  unsigned short* entb = (unsigned short*)alloc((size_t)B_ * N_ * E_ * 2);
  float* fn1 = (float*)alloc((size_t)B_ * F_ * 4);
  float* fn2 = (float*)alloc((size_t)B_ * F_ * 4);
  float* en  = (float*)alloc((size_t)B_ * N_ * 4);
  float* hop = (float*)alloc((size_t)4 * B_ * E_ * 4);
  float* c1  = (float*)alloc((size_t)2 * B_ * F_ * 4);
  float* c2  = (float*)alloc((size_t)2 * B_ * F_ * 4);
  float* s0  = (float*)alloc((size_t)B_ * F_ * 4);
  float* nsc = (float*)alloc((size_t)2 * B_ * N_ * 4);
  float* zs  = (float*)alloc((size_t)2 * B_ * 10 * 4);
  int*   zi  = (int*)  alloc((size_t)2 * B_ * 10 * 4);
  float* br  = (float*)alloc((size_t)2 * B_ * 10 * 4);

  k_hoprel<<<64, 256, 0, stream>>>(rel, W, bias, hop);
  k_cvt<<<B_ * F_, 256, 0, stream>>>(farg1, fa1b, fn1);
  k_cvt<<<B_ * F_, 256, 0, stream>>>(farg2, fa2b, fn2);
  k_cvt<<<B_ * N_, 256, 0, stream>>>(ent, entb, en);
  k_const<<<dim3(F_ / 8, B_), 256, 0, stream>>>(rel, arg1, arg2, frel, farg1, farg2,
                                                hop, nbf, c1, c2, s0);
  k_big<<<dim3(N_ / 64, B_, 2), 128, 0, stream>>>(entb, fa1b, fa2b, en, fn1, fn2, c1, nsc);
  k_topk<<<2 * B_, 256, 0, stream>>>(nsc, zs, zi);
  k_hop2<<<2 * B_ * 10, 256, 0, stream>>>(ent, farg1, farg2, c2, zs, zi, br);
  k_final<<<B_, 256, 0, stream>>>(s0, br, out);
}
